// PointNetPPEncDec_67010079752488
// MI455X (gfx1250) — matrix operand order repaired and checked by reading, not compiled, not run on hardware
//
#include <hip/hip_runtime.h>
#include <cstdint>
#include <cstddef>

// ---------------------------------------------------------------------------
// PointNet++ MSG part-seg forward, CDNA5 (gfx1250), wave32.
//
// All matmuls run on v_wmma_f32_16x16x32_f16 with branch-free 32-byte
// fragment loads and 2x2 (32x32) register blocking per wave:
//   - Weights pre-packed (once per launch) to f16 in B-fragment lane order.
//   - Activations kept in packed A-fragment order (f16) in LDS / workspace.
// SA branches are fully fused per-centroid in LDS (grouped activations never
// touch HBM), so the network is WMMA-bound instead of HBM-bound.
// ---------------------------------------------------------------------------

#define BN_INV 0.99999500003749968f   // 1/sqrt(1+1e-5)

typedef __attribute__((ext_vector_type(16))) _Float16 v16h;
typedef __attribute__((ext_vector_type(8)))  float    v8f;

struct LayerG { const _Float16* Wp; const float* b; const float* g; const float* be; };

// Packed-A slot for channel K (within its 32-chunk):
//   laneHi = (K>>3)&1 ; slot-in-half = (K&7) + ((K&16)>>1) ; q = laneHi*16 + slot
// Exact inverse of the ISA 16-bit A-matrix VGPR interleave, so a lane's 16
// halves [laneHi*16, laneHi*16+16) are one contiguous 32B load.
__device__ __forceinline__ int qof(int c) {
  return ((c >> 3) & 1) * 16 + (c & 7) + ((c >> 4) & 1) * 8;
}

#define WMMA_F16(A, B, C) \
  __builtin_amdgcn_wmma_f32_16x16x32_f16(false, (A), false, (B), (short)0, (C), false, false)

// Epilogue: bias + BN + relu, store one 16x16 tile.
// outPacked: packed-A f16 layout (zero pads N); else plain f32 [rows,Cout].
__device__ __forceinline__ void store_tile(
    const v8f& acc, int mbase, int n, int laneHi,
    const float* __restrict__ bias, const float* __restrict__ gam,
    const float* __restrict__ bet, void* __restrict__ out,
    int rows, int Cout, int outPacked)
{
  float bb  = (n < Cout) ? bias[n] : 0.f;
  float gg  = (n < Cout) ? gam[n]  : 0.f;
  float be2 = (n < Cout) ? bet[n]  : 0.f;
  if (outPacked) {
    _Float16* O = (_Float16*)out;
    int kcN = n >> 5, q = qof(n);
#pragma unroll
    for (int j = 0; j < 8; ++j) {
      int mm = mbase + j + (laneHi ? 8 : 0);
      float y = fmaxf((acc[j] + bb) * BN_INV * gg + be2, 0.f);
      O[((size_t)kcN * rows + mm) * 32 + q] = (_Float16)((n < Cout) ? y : 0.f);
    }
  } else if (n < Cout) {
    float* O = (float*)out;
#pragma unroll
    for (int j = 0; j < 8; ++j) {
      int mm = mbase + j + (laneHi ? 8 : 0);
      O[(size_t)mm * Cout + n] = fmaxf((acc[j] + bb) * BN_INV * gg + be2, 0.f);
    }
  }
}

// LDS variant (always packed f16, rows = nsample).
__device__ __forceinline__ void store_tile_lds(
    const v8f& acc, int mbase, int n, int laneHi,
    const float* __restrict__ bias, const float* __restrict__ gam,
    const float* __restrict__ bet, _Float16* __restrict__ dst,
    int nsample, int Cout)
{
  float bb  = (n < Cout) ? bias[n] : 0.f;
  float gg  = (n < Cout) ? gam[n]  : 0.f;
  float be2 = (n < Cout) ? bet[n]  : 0.f;
  int kcN = n >> 5, q = qof(n);
#pragma unroll
  for (int j = 0; j < 8; ++j) {
    int mm = mbase + j + (laneHi ? 8 : 0);
    float y = fmaxf((acc[j] + bb) * BN_INV * gg + be2, 0.f);
    dst[((size_t)kcN * nsample + mm) * 32 + q] = (_Float16)((n < Cout) ? y : 0.f);
  }
}

// ---------------------------------------------------------------------------
// Weight pack: W[Cin,Cout] f32 -> Wp[kc][n][q] f16, K/N zero-padded to 32.
// B-fragment order matches packed-A: q = qof(K%32) (same interleave as A).
// ---------------------------------------------------------------------------
__global__ void pack_w_kernel(const float* __restrict__ W, int Cin, int Cout,
                              _Float16* __restrict__ Wp, int Npad, int total)
{
  int t = blockIdx.x * blockDim.x + threadIdx.x;
  if (t >= total) return;
  int q = t & 31;
  int rest = t >> 5;
  int n = rest % Npad;
  int kc = rest / Npad;
  int K = kc * 32 + (q & 7) + ((q >> 4) & 1) * 8 + ((q >> 3) & 1) * 16;
  Wp[t] = (K < Cin && n < Cout) ? (_Float16)W[(size_t)K * Cout + n] : (_Float16)0.f;
}

// ---------------------------------------------------------------------------
// Generic dense layer on packed-A input. One wave per 32x32 block (2x2 WMMA
// tiles): 4 fragment loads -> 4 WMMAs per K-chunk (2x operand reuse).
// ---------------------------------------------------------------------------
__global__ __launch_bounds__(128) void layer_kernel(
    const _Float16* __restrict__ Xp, const _Float16* __restrict__ Wp,
    const float* __restrict__ bias, const float* __restrict__ gam,
    const float* __restrict__ bet, void* __restrict__ out,
    int rows, int Kc, int Cout, int Npad, int outPacked)
{
  int lane = threadIdx.x & 31;
  int laneHi = (lane >> 4) & 1;
  int tilesN = Npad >> 5;
  int tiles = (rows >> 5) * tilesN;
  int gw = (blockDim.x >> 5) * gridDim.x;
  const v16h* Av = (const v16h*)Xp;
  const v16h* Bv = (const v16h*)Wp;
  for (int t = blockIdx.x * (blockDim.x >> 5) + (threadIdx.x >> 5); t < tiles; t += gw) {
    int tm = t / tilesN, tn = t - tm * tilesN;
    int n0 = (tn << 5) + (lane & 15);
    v8f c00 = {0.f,0.f,0.f,0.f,0.f,0.f,0.f,0.f};
    v8f c01 = c00, c10 = c00, c11 = c00;
    size_t a0 = ((size_t)(tm << 5) + (lane & 15)) * 2 + laneHi;
    size_t a1 = a0 + 32;                         // +16 rows
    size_t b0 = (size_t)n0 * 2 + laneHi;
    size_t b1 = b0 + 32;                         // +16 cols
    size_t astep = (size_t)rows * 2, bstep = (size_t)Npad * 2;
    for (int kc = 0; kc < Kc; ++kc) {
      __builtin_prefetch(Av + a0 + astep);       // global_prefetch_b8
      __builtin_prefetch(Bv + b0 + bstep);
      v16h A0 = Av[a0], A1 = Av[a1];
      v16h B0 = Bv[b0], B1 = Bv[b1];
      c00 = WMMA_F16(A0, B0, c00);
      c01 = WMMA_F16(A0, B1, c01);
      c10 = WMMA_F16(A1, B0, c10);
      c11 = WMMA_F16(A1, B1, c11);
      a0 += astep; a1 += astep; b0 += bstep; b1 += bstep;
    }
    int mb = tm << 5;
    store_tile(c00, mb,      n0,      laneHi, bias, gam, bet, out, rows, Cout, outPacked);
    store_tile(c01, mb,      n0 + 16, laneHi, bias, gam, bet, out, rows, Cout, outPacked);
    store_tile(c10, mb + 16, n0,      laneHi, bias, gam, bet, out, rows, Cout, outPacked);
    store_tile(c11, mb + 16, n0 + 16, laneHi, bias, gam, bet, out, rows, Cout, outPacked);
  }
}

// ---------------------------------------------------------------------------
// Fused SA layer: packed-A LDS -> packed-A LDS, 32x32 blocking, packed weights.
// ---------------------------------------------------------------------------
__device__ __forceinline__ void fused_layer_packed(
    const _Float16* __restrict__ Wp, const float* __restrict__ bias,
    const float* __restrict__ gam, const float* __restrict__ bet,
    int Kc, int Cout, int Npad,
    const _Float16* __restrict__ src, _Float16* __restrict__ dst, int nsample)
{
  int lane = threadIdx.x & 31;
  int laneHi = (lane >> 4) & 1;
  int wave = threadIdx.x >> 5, nw = blockDim.x >> 5;
  int tilesM = nsample >> 5, tilesN = Npad >> 5;
  const v16h* Bv = (const v16h*)Wp;
  const v16h* Sv = (const v16h*)src;
  for (int t = wave; t < tilesM * tilesN; t += nw) {
    int tm = t / tilesN, tn = t - tm * tilesN;
    int n0 = (tn << 5) + (lane & 15);
    v8f c00 = {0.f,0.f,0.f,0.f,0.f,0.f,0.f,0.f};
    v8f c01 = c00, c10 = c00, c11 = c00;
    size_t a0 = ((size_t)(tm << 5) + (lane & 15)) * 2 + laneHi;
    size_t a1 = a0 + 32;
    size_t b0 = (size_t)n0 * 2 + laneHi;
    size_t b1 = b0 + 32;
    size_t astep = (size_t)nsample * 2, bstep = (size_t)Npad * 2;
    for (int kc = 0; kc < Kc; ++kc) {
      v16h A0 = Sv[a0], A1 = Sv[a1];
      v16h B0 = Bv[b0], B1 = Bv[b1];
      c00 = WMMA_F16(A0, B0, c00);
      c01 = WMMA_F16(A0, B1, c01);
      c10 = WMMA_F16(A1, B0, c10);
      c11 = WMMA_F16(A1, B1, c11);
      a0 += astep; a1 += astep; b0 += bstep; b1 += bstep;
    }
    int mb = tm << 5;
    store_tile_lds(c00, mb,      n0,      laneHi, bias, gam, bet, dst, nsample, Cout);
    store_tile_lds(c01, mb,      n0 + 16, laneHi, bias, gam, bet, dst, nsample, Cout);
    store_tile_lds(c10, mb + 16, n0,      laneHi, bias, gam, bet, dst, nsample, Cout);
    store_tile_lds(c11, mb + 16, n0 + 16, laneHi, bias, gam, bet, dst, nsample, Cout);
  }
}

// ---------------------------------------------------------------------------
// Fused SA branch: gather(group) -> 3x WMMA MLP (LDS resident) -> max-pool.
// One workgroup (256 thr = 8 waves) per centroid.
// ---------------------------------------------------------------------------
__global__ __launch_bounds__(256) void sa_fused_kernel(
    const float* __restrict__ xyz, int N,
    const float* __restrict__ points, int Cpts,
    const float* __restrict__ new_xyz, const int* __restrict__ gidx,
    int S, int nsample,
    LayerG L0, LayerG L1, LayerG L2,
    int c0, int c1, int c2, int c3,
    float* __restrict__ out, int outStride, int chanOff, int buf1Off)
{
  extern __shared__ _Float16 ldsh[];
  _Float16* buf0 = ldsh;
  _Float16* buf1 = ldsh + buf1Off;
  int b = blockIdx.x / S, s = blockIdx.x - b * S;
  const int* g = gidx + (size_t)(b * S + s) * nsample;

  // Stage grouped input (f32 gather -> packed f16 A layout), zero-padding K.
  int Kc0 = (c0 + 31) >> 5;
  int c0pad = Kc0 * 32;
  for (int t = threadIdx.x; t < nsample * c0pad; t += blockDim.x) {
    int k = t / c0pad, c = t - k * c0pad;
    float v = 0.f;
    if (c < c0) {
      int gi = g[k];
      if (c < Cpts) {
        v = points[((size_t)(b * N + gi)) * Cpts + c];
      } else {
        int d = c - Cpts;
        v = xyz[(b * N + gi) * 3 + d] - new_xyz[(b * S + s) * 3 + d];
      }
    }
    buf0[((size_t)((c >> 5) * nsample + k)) * 32 + qof(c)] = (_Float16)v;
  }
  __syncthreads();

  fused_layer_packed(L0.Wp, L0.b, L0.g, L0.be, Kc0, c1, (c1 + 31) & ~31,
                     buf0, buf1, nsample);
  __syncthreads();
  fused_layer_packed(L1.Wp, L1.b, L1.g, L1.be, (c1 + 31) >> 5, c2, (c2 + 31) & ~31,
                     buf1, buf0, nsample);
  __syncthreads();
  fused_layer_packed(L2.Wp, L2.b, L2.g, L2.be, (c2 + 31) >> 5, c3, (c3 + 31) & ~31,
                     buf0, buf1, nsample);
  __syncthreads();

  // Max-pool over nsample neighbors (reads packed f16, writes plain f32).
  for (int c = threadIdx.x; c < c3; c += blockDim.x) {
    int kc = c >> 5, q = qof(c);
    float mx = (float)buf1[((size_t)kc * nsample) * 32 + q];
    for (int k = 1; k < nsample; ++k)
      mx = fmaxf(mx, (float)buf1[((size_t)kc * nsample + k) * 32 + q]);
    out[(size_t)(b * S + s) * outStride + chanOff + c] = mx;
  }
}

// ---------------------------------------------------------------------------
// Farthest point sampling: one workgroup per batch, dist[] resident in LDS.
// ---------------------------------------------------------------------------
__global__ __launch_bounds__(256) void fps_kernel(
    const float* __restrict__ xyz, int N, int S,
    int* __restrict__ fidx, float* __restrict__ new_xyz)
{
  extern __shared__ float lds[];
  float* dist = lds;                 // N
  float* rmax = lds + N;             // 256
  int*   rarg = (int*)(rmax + 256);  // 256
  __shared__ int s_far;
  int b = blockIdx.x;
  for (int i = threadIdx.x; i < N; i += blockDim.x) dist[i] = 1e10f;
  if (threadIdx.x == 0) s_far = 0;
  __syncthreads();
  for (int it = 0; it < S; ++it) {
    int far = s_far;
    if (threadIdx.x == 0) fidx[b * S + it] = far;
    if (threadIdx.x < 3)
      new_xyz[(b * S + it) * 3 + threadIdx.x] = xyz[(b * N + far) * 3 + threadIdx.x];
    float cx = xyz[(b * N + far) * 3 + 0];
    float cy = xyz[(b * N + far) * 3 + 1];
    float cz = xyz[(b * N + far) * 3 + 2];
    float best = -1.f; int bidx = 0;
    for (int i = threadIdx.x; i < N; i += blockDim.x) {
      float dx = xyz[(b * N + i) * 3 + 0] - cx;
      float dy = xyz[(b * N + i) * 3 + 1] - cy;
      float dz = xyz[(b * N + i) * 3 + 2] - cz;
      float d  = dx * dx + dy * dy + dz * dz;
      float nd = fminf(dist[i], d);
      dist[i] = nd;
      if (nd > best) { best = nd; bidx = i; }
    }
    rmax[threadIdx.x] = best; rarg[threadIdx.x] = bidx;
    __syncthreads();
    for (int o = 128; o > 0; o >>= 1) {
      if (threadIdx.x < o) {
        float ov = rmax[threadIdx.x + o];
        if (ov > rmax[threadIdx.x] ||
            (ov == rmax[threadIdx.x] && rarg[threadIdx.x + o] < rarg[threadIdx.x])) {
          rmax[threadIdx.x] = ov; rarg[threadIdx.x] = rarg[threadIdx.x + o];
        }
      }
      __syncthreads();
    }
    if (threadIdx.x == 0) s_far = rarg[0];
    __syncthreads();
  }
}

// Ball query: first nsample in-range indices (index order), pad with first.
__global__ void ball_query_kernel(
    const float* __restrict__ xyz, int N, const float* __restrict__ nx,
    float r2, int nsample, int* __restrict__ gidx, int total)
{
  int t = blockIdx.x * blockDim.x + threadIdx.x;
  if (t >= total) return;
  int S = total >> 3;          // total = 8*S
  int bb = t / S;
  float cx = nx[t * 3 + 0], cy = nx[t * 3 + 1], cz = nx[t * 3 + 2];
  int cnt = 0, first = -1;
  int* g = gidx + (size_t)t * nsample;
  for (int i = 0; i < N && cnt < nsample; ++i) {
    float dx = xyz[(bb * N + i) * 3 + 0] - cx;
    float dy = xyz[(bb * N + i) * 3 + 1] - cy;
    float dz = xyz[(bb * N + i) * 3 + 2] - cz;
    if (dx * dx + dy * dy + dz * dz <= r2) {
      if (first < 0) first = i;
      g[cnt++] = i;
    }
  }
  if (first < 0) first = N - 1;
  for (; cnt < nsample; ++cnt) g[cnt] = first;
}

// 3-NN with inverse-distance weights.
__global__ void knn3_kernel(
    const float* __restrict__ x1, int n1, const float* __restrict__ x2, int n2,
    int* __restrict__ idx, float* __restrict__ w, int total)
{
  int t = blockIdx.x * blockDim.x + threadIdx.x;
  if (t >= total) return;
  int b = t / n1;
  float px = x1[t * 3 + 0], py = x1[t * 3 + 1], pz = x1[t * 3 + 2];
  float d0 = 1e30f, d1 = 1e30f, d2 = 1e30f;
  int i0 = 0, i1 = 0, i2 = 0;
  for (int j = 0; j < n2; ++j) {
    float dx = x2[(b * n2 + j) * 3 + 0] - px;
    float dy = x2[(b * n2 + j) * 3 + 1] - py;
    float dz = x2[(b * n2 + j) * 3 + 2] - pz;
    float d = dx * dx + dy * dy + dz * dz;
    if (d < d0)      { d2 = d1; i2 = i1; d1 = d0; i1 = i0; d0 = d; i0 = j; }
    else if (d < d1) { d2 = d1; i2 = i1; d1 = d; i1 = j; }
    else if (d < d2) { d2 = d; i2 = j; }
  }
  float w0 = 1.f / (d0 + 1e-8f), w1 = 1.f / (d1 + 1e-8f), w2 = 1.f / (d2 + 1e-8f);
  float inv = 1.f / (w0 + w1 + w2);
  idx[t * 3 + 0] = i0; idx[t * 3 + 1] = i1; idx[t * 3 + 2] = i2;
  w[t * 3 + 0] = w0 * inv; w[t * 3 + 1] = w1 * inv; w[t * 3 + 2] = w2 * inv;
}

// concat(points1, interp3(points2)) -> packed-A f16 [kc][row][32], K zero-pad.
__global__ void interp_concat_kernel(
    const float* __restrict__ p1, int C1, const float* __restrict__ p2, int C2,
    const int* __restrict__ idx, const float* __restrict__ w,
    _Float16* __restrict__ Xp, int n1, int n2, int rows, int Cpad, int total)
{
  int t = blockIdx.x * blockDim.x + threadIdx.x;
  if (t >= total) return;
  int row = t / Cpad, c = t - row * Cpad;
  int b = row / n1;
  float v = 0.f;
  if (c < C1) {
    v = p1[(size_t)row * C1 + c];
  } else if (c < C1 + C2) {
    int cc = c - C1;
    const int* id = idx + (size_t)row * 3;
    const float* ww = w + (size_t)row * 3;
    float acc = 0.f;
#pragma unroll
    for (int k = 0; k < 3; ++k)
      acc += ww[k] * p2[((size_t)b * n2 + id[k]) * C2 + cc];
    v = acc;
  }
  Xp[((size_t)(c >> 5) * rows + row) * 32 + qof(c)] = (_Float16)v;
}

// sa3 input: concat(l2_xyz, l2_points) -> packed [B*128, pad32(515)=544]
__global__ void sa3_concat_kernel(
    const float* __restrict__ nx2, const float* __restrict__ l2p,
    _Float16* __restrict__ Xp, int rows, int Cpad, int total)
{
  int t = blockIdx.x * blockDim.x + threadIdx.x;
  if (t >= total) return;
  int row = t / Cpad, c = t - row * Cpad;
  float v = 0.f;
  if (c < 3) v = nx2[row * 3 + c];
  else if (c < 515) v = l2p[(size_t)row * 512 + (c - 3)];
  Xp[((size_t)(c >> 5) * rows + row) * 32 + qof(c)] = (_Float16)v;
}

// fp3 input: concat(l2_points, broadcast(l3max)) -> packed [B*128, 1536]
__global__ void fp3_concat_kernel(
    const float* __restrict__ l2p, const float* __restrict__ l3,
    _Float16* __restrict__ Xp, int rows, int Cpad, int total)
{
  int t = blockIdx.x * blockDim.x + threadIdx.x;
  if (t >= total) return;
  int row = t / Cpad, c = t - row * Cpad;
  int b = row >> 7; // /128
  float v = (c < 512) ? l2p[(size_t)row * 512 + c] : l3[(size_t)b * 1024 + (c - 512)];
  Xp[((size_t)(c >> 5) * rows + row) * 32 + qof(c)] = (_Float16)v;
}

// max over point axis on packed-A input: Xp rows = B*P -> out[B,C] (f32)
__global__ void maxpool_pts_kernel(
    const _Float16* __restrict__ Xp, float* __restrict__ out,
    int P, int C, int rows, int total)
{
  int t = blockIdx.x * blockDim.x + threadIdx.x;
  if (t >= total) return;
  int b = t / C, c = t - b * C;
  int kc = c >> 5, q = qof(c);
  float mx = -1e30f;
  for (int p = 0; p < P; ++p)
    mx = fmaxf(mx, (float)Xp[((size_t)kc * rows + b * P + p) * 32 + q]);
  out[t] = mx;
}

// head: out[b,c,n] = feat[b,n,:] . W[:,c] + bias[c]   (transposed store)
__global__ void head_kernel(
    const float* __restrict__ feat, const float* __restrict__ W,
    const float* __restrict__ bias, float* __restrict__ out, int P, int total)
{
  int t = blockIdx.x * blockDim.x + threadIdx.x;
  if (t >= total) return;
  int n = t & 4095;
  int rest = t >> 12;
  int c = rest % P, b = rest / P;
  const float* f = feat + ((size_t)b * 4096 + n) * 128;
  float acc = bias[c];
  for (int k = 0; k < 128; ++k) acc += f[k] * W[k * P + c];
  out[t] = acc;
}

// ---------------------------------------------------------------------------
// Host orchestration
// ---------------------------------------------------------------------------
extern "C" void kernel_launch(void* const* d_in, const int* in_sizes, int n_in,
                              void* d_out, int out_size, void* d_ws, size_t ws_size,
                              hipStream_t stream)
{
  (void)in_sizes; (void)n_in; (void)out_size; (void)ws_size;
  const int Bb = 8, Nn = 4096, S1 = 512, S2 = 128;
  const float* xyz = (const float*)d_in[0];

  // Parameters in setup_inputs() insertion order; each layer: W,b,gamma,beta.
  struct RawL { const float *W, *b, *g, *be; };
  int cur = 1;
  auto take = [&](RawL& L) {
    L.W  = (const float*)d_in[cur++];
    L.b  = (const float*)d_in[cur++];
    L.g  = (const float*)d_in[cur++];
    L.be = (const float*)d_in[cur++];
  };
  RawL sa1r[3][3], sa2r[2][3], sa3r[3], fp3r[2], fp2r[2], fp1r[2], conv1r;
  for (int br = 0; br < 3; ++br) for (int l = 0; l < 3; ++l) take(sa1r[br][l]);
  for (int br = 0; br < 2; ++br) for (int l = 0; l < 3; ++l) take(sa2r[br][l]);
  for (int l = 0; l < 3; ++l) take(sa3r[l]);
  for (int l = 0; l < 2; ++l) take(fp3r[l]);
  for (int l = 0; l < 2; ++l) take(fp2r[l]);
  for (int l = 0; l < 2; ++l) take(fp1r[l]);
  take(conv1r);
  const float *hW[16], *hB[16];
  for (int h = 0; h < 16; ++h) {
    hW[h] = (const float*)d_in[cur++];
    hB[h] = (const float*)d_in[cur++];
  }

  // Workspace carve (256-aligned).
  char* ws = (char*)d_ws;
  size_t off = 0;
  auto carveB = [&](size_t bytes) -> void* {
    void* p = ws + off;
    off += (bytes + 255) & ~(size_t)255;
    return p;
  };
  auto carve = [&](size_t elems) -> void* { return carveB(elems * 4); };
  int*   fidx1 = (int*)carve((size_t)Bb * S1);
  float* nx1   = (float*)carve((size_t)Bb * S1 * 3);
  int*   gidx  = (int*)carve((size_t)Bb * S1 * 128);
  int*   fidx2 = (int*)carve((size_t)Bb * S2);
  float* nx2   = (float*)carve((size_t)Bb * S2 * 3);
  float* l1p   = (float*)carve((size_t)Bb * S1 * 320);
  float* l2p   = (float*)carve((size_t)Bb * S2 * 512);
  void*  GA    = carve(4718592);   // generic buffers (f32 plain or f16 packed)
  void*  GB    = carve(4718592);
  float* l3mx  = (float*)carve((size_t)Bb * 1024);
  float* l2pn  = (float*)carve((size_t)Bb * S2 * 256);
  float* l1pn  = (float*)carve((size_t)Bb * S1 * 128);
  int*   kidx  = (int*)carve((size_t)Bb * Nn * 3);
  float* kw    = (float*)carve((size_t)Bb * Nn * 3);

  // Pre-pack all WMMA-layer weights to f16 B-fragment layout.
  auto packW = [&](const RawL& R, int Cin, int Cout) -> LayerG {
    int Kc = (Cin + 31) >> 5;
    int Npad = (Cout + 31) & ~31;
    int total = Kc * 32 * Npad;
    _Float16* Wp = (_Float16*)carveB((size_t)total * 2);
    pack_w_kernel<<<(total + 255) / 256, 256, 0, stream>>>(R.W, Cin, Cout, Wp, Npad, total);
    return LayerG{Wp, R.b, R.g, R.be};
  };

  const int ch1[3][4] = {{3,32,32,64},{3,64,64,128},{3,64,96,128}};
  const int ch2[2][4] = {{323,128,128,256},{323,128,196,256}};
  LayerG sa1g[3][3], sa2g[2][3], sa3g[3], fp3g[2], fp2g[2], fp1g[2], conv1g;
  for (int br = 0; br < 3; ++br)
    for (int l = 0; l < 3; ++l)
      sa1g[br][l] = packW(sa1r[br][l], ch1[br][l], ch1[br][l + 1]);
  for (int br = 0; br < 2; ++br)
    for (int l = 0; l < 3; ++l)
      sa2g[br][l] = packW(sa2r[br][l], ch2[br][l], ch2[br][l + 1]);
  {
    const int c[4] = {515, 256, 512, 1024};
    for (int l = 0; l < 3; ++l) sa3g[l] = packW(sa3r[l], c[l], c[l + 1]);
  }
  fp3g[0] = packW(fp3r[0], 1536, 256); fp3g[1] = packW(fp3r[1], 256, 256);
  fp2g[0] = packW(fp2r[0], 576, 256);  fp2g[1] = packW(fp2r[1], 256, 128);
  fp1g[0] = packW(fp1r[0], 131, 128);  fp1g[1] = packW(fp1r[1], 128, 128);
  conv1g  = packW(conv1r, 128, 128);

  auto layerL = [&](const void* Xp, const LayerG& L, void* out,
                    int rows, int Cin, int Cout, int outPacked) {
    int Kc = (Cin + 31) >> 5;
    int Npad = (Cout + 31) & ~31;
    int tiles = (rows >> 5) * (Npad >> 5);    // 32x32 blocks
    int blocks = (tiles + 3) / 4;
    layer_kernel<<<blocks, 128, 0, stream>>>((const _Float16*)Xp, L.Wp,
        L.b, L.g, L.be, out, rows, Kc, Cout, Npad, outPacked);
  };
  auto saL = [&](const float* pxyz, int N, const float* pts, int Cpts,
                 const float* nx, int S, int ns, const LayerG* L, const int* ch,
                 float* out, int outStride, int chanOff) {
    int Kc0 = (ch[0] + 31) >> 5, Kc1 = (ch[1] + 31) >> 5;
    int Kc2 = (ch[2] + 31) >> 5, Kc3 = (ch[3] + 31) >> 5;
    int H0 = Kc0 > Kc2 ? Kc0 : Kc2;
    int H1 = Kc1 > Kc3 ? Kc1 : Kc3;
    int buf1Off = ns * 32 * H0;                       // halves
    size_t shb = (size_t)ns * 32 * (H0 + H1) * 2;     // bytes
    sa_fused_kernel<<<Bb * S, 256, shb, stream>>>(
        pxyz, N, pts, Cpts, nx, gidx, S, ns, L[0], L[1], L[2],
        ch[0], ch[1], ch[2], ch[3], out, outStride, chanOff, buf1Off);
  };

  // ---- SA1 (512 centroids, 3 scales, cin=3, concat out 320ch) --------------
  fps_kernel<<<Bb, 256, (size_t)(Nn + 512) * 4, stream>>>(xyz, Nn, S1, fidx1, nx1);
  {
    const int ns[3] = {32, 64, 128};
    const float r2[3] = {0.01f, 0.04f, 0.16f};
    const int coff[3] = {0, 64, 192};
    for (int sc = 0; sc < 3; ++sc) {
      ball_query_kernel<<<(Bb * S1 + 255) / 256, 256, 0, stream>>>(
          xyz, Nn, nx1, r2[sc], ns[sc], gidx, Bb * S1);
      saL(xyz, Nn, nullptr, 0, nx1, S1, ns[sc], sa1g[sc], ch1[sc], l1p, 320, coff[sc]);
    }
  }

  // ---- SA2 (128 centroids, 2 scales, cin=323, concat out 512ch) ------------
  fps_kernel<<<Bb, 256, (size_t)(S1 + 512) * 4, stream>>>(nx1, S1, S2, fidx2, nx2);
  {
    const int ns[2] = {64, 128};
    const float r2[2] = {0.16f, 0.64f};
    const int coff[2] = {0, 256};
    for (int sc = 0; sc < 2; ++sc) {
      ball_query_kernel<<<(Bb * S2 + 255) / 256, 256, 0, stream>>>(
          nx1, S1, nx2, r2[sc], ns[sc], gidx, Bb * S2);
      saL(nx1, S1, l1p, 320, nx2, S2, ns[sc], sa2g[sc], ch2[sc], l2p, 512, coff[sc]);
    }
  }

  // ---- SA3 global: [B*128, 515] -> 256 -> 512 -> 1024 -> max over 128 ------
  {
    int rows = Bb * S2, Cpad = 544, tot = rows * Cpad;
    sa3_concat_kernel<<<(tot + 255) / 256, 256, 0, stream>>>(
        nx2, l2p, (_Float16*)GA, rows, Cpad, tot);
    layerL(GA, sa3g[0], GB, rows, 515, 256, 1);
    layerL(GB, sa3g[1], GA, rows, 256, 512, 1);
    layerL(GA, sa3g[2], GB, rows, 512, 1024, 1);
    maxpool_pts_kernel<<<(Bb * 1024 + 255) / 256, 256, 0, stream>>>(
        (const _Float16*)GB, l3mx, S2, 1024, rows, Bb * 1024);
  }

  // ---- FP3: concat(l2p, bcast l3) [B*128,1536] -> 256 -> 256 ---------------
  {
    int rows = Bb * S2, Cpad = 1536, tot = rows * Cpad;
    fp3_concat_kernel<<<(tot + 255) / 256, 256, 0, stream>>>(
        l2p, l3mx, (_Float16*)GA, rows, Cpad, tot);
    layerL(GA, fp3g[0], GB, rows, 1536, 256, 1);
    layerL(GB, fp3g[1], l2pn, rows, 256, 256, 0);
  }

  // ---- FP2: 3-NN(l1_xyz <- l2_xyz), concat(l1p, interp) [B*512,576] --------
  {
    knn3_kernel<<<(Bb * S1 + 255) / 256, 256, 0, stream>>>(
        nx1, S1, nx2, S2, kidx, kw, Bb * S1);
    int rows = Bb * S1, Cpad = 576, tot = rows * Cpad;
    interp_concat_kernel<<<(tot + 255) / 256, 256, 0, stream>>>(
        l1p, 320, l2pn, 256, kidx, kw, (_Float16*)GA, S1, S2, rows, Cpad, tot);
    layerL(GA, fp2g[0], GB, rows, 576, 256, 1);
    layerL(GB, fp2g[1], l1pn, rows, 256, 128, 0);
  }

  // ---- FP1: 3-NN(xyz <- l1_xyz), concat(xyz, interp) [B*4096,131] ----------
  {
    knn3_kernel<<<(Bb * Nn + 255) / 256, 256, 0, stream>>>(
        xyz, Nn, nx1, S1, kidx, kw, Bb * Nn);
    int rows = Bb * Nn, Cpad = 160, tot = rows * Cpad;
    interp_concat_kernel<<<(tot + 255) / 256, 256, 0, stream>>>(
        xyz, 3, l1pn, 128, kidx, kw, (_Float16*)GA, Nn, S1, rows, Cpad, tot);
    layerL(GA, fp1g[0], GB, rows, 131, 128, 1);
    layerL(GB, fp1g[1], GA, rows, 128, 128, 1);
    // conv1 (plain f32 out for the heads)
    layerL(GA, conv1g, GB, rows, 128, 128, 0);
  }

  // ---- 16 heads ------------------------------------------------------------
  {
    const int parts[16] = {4,2,2,4,4,3,3,2,4,2,6,2,3,3,3,3};
    float* outp = (float*)d_out;
    size_t oo = 0;
    for (int h = 0; h < 16; ++h) {
      int P = parts[h];
      int total = Bb * P * Nn;
      head_kernel<<<(total + 255) / 256, 256, 0, stream>>>(
          (const float*)GB, hW[h], hB[h], outp + oo, P, total);
      oo += (size_t)total;
    }
  }
}
